// Net_3607772528718
// MI455X (gfx1250) — matrix operand order repaired and checked by reading, not compiled, not run on hardware
//
#include <hip/hip_runtime.h>
#include <hip/hip_bf16.h>

typedef __attribute__((ext_vector_type(16))) _Float16 v16h;
typedef __attribute__((ext_vector_type(8)))  _Float16 v8h;
typedef __attribute__((ext_vector_type(8)))  float    v8f;

#define NEG_SLOPE 0.2f

// ---------- ordered-uint encoding for float atomic max ----------
static __device__ __forceinline__ unsigned f2ord(float f) {
  unsigned b = __float_as_uint(f);
  return (b & 0x80000000u) ? ~b : (b | 0x80000000u);
}
static __device__ __forceinline__ float ord2f(unsigned u) {
  unsigned b = (u & 0x80000000u) ? (u & 0x7FFFFFFFu) : ~u;
  return __uint_as_float(b);
}
static __device__ __forceinline__ float lrelu(float x) { return x > 0.f ? x : NEG_SLOPE * x; }

// ---------- utility kernels ----------
__global__ void k_fill_u32(unsigned* __restrict__ p, unsigned v, long n) {
  long i = blockIdx.x * (long)blockDim.x + threadIdx.x;
  if (i < n) p[i] = v;
}

__global__ void k_cvt_f16(const float* __restrict__ s, _Float16* __restrict__ d, long n) {
  long i = blockIdx.x * (long)blockDim.x + threadIdx.x;
  if (i < n) d[i] = (_Float16)s[i];
}

// W [K,N] f32 -> Wt [N,K] f16 (each output column becomes a contiguous K-run)
__global__ void k_wt(const float* __restrict__ W, _Float16* __restrict__ Wt, int K, int N) {
  int i = blockIdx.x * blockDim.x + threadIdx.x;
  if (i >= K * N) return;
  int k = i / N, n = i - k * N;
  Wt[(long)n * K + k] = (_Float16)W[i];
}

// ---------- WMMA GEMM, 16xN strip per wave ----------
// C[M,N] = A[M,K] * B[K,N]; A f16 row-major, B given as Bt[N,K] f16.
// Each wave owns one 16-row tile of A and all NT = N/16 column tiles:
// per K-step the A fragment is loaded once and reused by NT WMMAs with NT
// independent accumulators (hides WMMA->WMMA hazards, cuts A traffic NTx).
template <int NT>
__global__ void k_gemm_strip(const _Float16* __restrict__ A, const _Float16* __restrict__ Bt,
                             float* __restrict__ C, int Mt /*#16-row tiles*/, int K) {
  constexpr int N = NT * 16;
  int wave = blockIdx.x * (blockDim.x >> 5) + (threadIdx.x >> 5);
  int lane = threadIdx.x & 31;
  if (wave >= Mt) return;

  long m = (long)wave * 16 + (lane & 15);
  // A fragment as loaded here (contiguous k, same as B): lanes 0-15 hold K=0..15; lanes 16-31 hold K=16..31
  int kbA = (lane < 16) ? 0 : 16;
  // B fragment as loaded here (same k order as A): lanes 0-15 hold K=0..15; lanes 16-31 hold K=16..31 (contiguous in Bt row)
  int kbB = (lane < 16) ? 0 : 16;

  const _Float16* arow = A + m * K;
  const _Float16* brow[NT];
#pragma unroll
  for (int t = 0; t < NT; ++t) brow[t] = Bt + ((long)(t * 16 + (lane & 15))) * K;

  v8f acc[NT];
#pragma unroll
  for (int t = 0; t < NT; ++t) acc[t] = {0.f, 0.f, 0.f, 0.f, 0.f, 0.f, 0.f, 0.f};

  for (int k0 = 0; k0 < K; k0 += 32) {
    v16h a = *(const v16h*)(arow + k0 + kbA);
#pragma unroll
    for (int t = 0; t < NT; ++t) {
      v16h b = *(const v16h*)(brow[t] + k0 + kbB);
      acc[t] = __builtin_amdgcn_wmma_f32_16x16x32_f16(false, a, false, b, (short)0,
                                                      acc[t], false, false);
    }
  }
  // C/D layout: lane column n; VGPR r -> row r (lanes 0-15) or r+8 (lanes 16-31)
  long mr = (long)wave * 16 + ((lane < 16) ? 0 : 8);
  int n = lane & 15;
#pragma unroll
  for (int t = 0; t < NT; ++t)
#pragma unroll
    for (int r = 0; r < 8; ++r)
      C[(mr + r) * N + t * 16 + n] = acc[t][r];
}

// ---------- attention coefficients: es/ed [N,H] = sum_c h[n,h,c]*a[h,c] ----------
__global__ void k_attn_coef(const float* __restrict__ h, const float* __restrict__ asrc,
                            const float* __restrict__ adst, float* __restrict__ es,
                            float* __restrict__ ed, int Nn, int H, int C) {
  int i = blockIdx.x * blockDim.x + threadIdx.x;
  if (i >= Nn * H) return;
  int n = i / H, hh = i - n * H;
  const float* row = h + (long)n * H * C + (long)hh * C;
  float s = 0.f, d = 0.f;
  for (int c = 0; c < C; ++c) {
    float v = row[c];
    s = fmaf(v, asrc[hh * C + c], s);
    d = fmaf(v, adst[hh * C + c], d);
  }
  es[i] = s;
  ed[i] = d;
}

// ---------- edge passes (self-loops appended after the E real edges) ----------
__global__ void k_edge_max(const int* __restrict__ ei, int E, int Nn,
                           const float* __restrict__ es, const float* __restrict__ ed,
                           unsigned* __restrict__ lmax, int H) {
  long t = blockIdx.x * (long)blockDim.x + threadIdx.x;
  long ET = (long)E + Nn;
  if (t >= ET * H) return;
  long e = t / H;
  int hh = (int)(t - e * H);
  int s, d;
  if (e < E) { s = ei[e]; d = ei[(long)E + e]; } else { s = d = (int)(e - E); }
  float lg = lrelu(es[(long)s * H + hh] + ed[(long)d * H + hh]);
  atomicMax(&lmax[(long)d * H + hh], f2ord(lg));
}

__global__ void k_edge_sum(const int* __restrict__ ei, int E, int Nn,
                           const float* __restrict__ es, const float* __restrict__ ed,
                           const unsigned* __restrict__ lmax, float* __restrict__ denom, int H) {
  long t = blockIdx.x * (long)blockDim.x + threadIdx.x;
  long ET = (long)E + Nn;
  if (t >= ET * H) return;
  long e = t / H;
  int hh = (int)(t - e * H);
  int s, d;
  if (e < E) { s = ei[e]; d = ei[(long)E + e]; } else { s = d = (int)(e - E); }
  long dh = (long)d * H + hh;
  __builtin_prefetch(&denom[dh], 1, 0);   // global_prefetch_b8 hint for the atomic line
  float lg = lrelu(es[(long)s * H + hh] + ed[dh]);
  atomicAdd(&denom[dh], __expf(lg - ord2f(lmax[dh])));
}

// one thread = (edge, head, 4 channels): float4 gather of h[src], scaled scatter-add to agg[dst]
__global__ void k_edge_agg(const int* __restrict__ ei, int E, int Nn,
                           const float* __restrict__ es, const float* __restrict__ ed,
                           const unsigned* __restrict__ lmax, const float* __restrict__ denom,
                           const float* __restrict__ h, float* __restrict__ agg, int H, int C) {
  long t = blockIdx.x * (long)blockDim.x + threadIdx.x;
  int Q = (H * C) >> 2;
  long ET = (long)E + Nn;
  if (t >= ET * Q) return;
  long e = t / Q;
  int q = (int)(t - e * Q);
  int c4 = q << 2;
  int hh = c4 / C;
  int c0 = c4 - hh * C;
  int s, d;
  if (e < E) { s = ei[e]; d = ei[(long)E + e]; } else { s = d = (int)(e - E); }
  long dh = (long)d * H + hh;
  float lg = lrelu(es[(long)s * H + hh] + ed[dh]);
  float alpha = __expf(lg - ord2f(lmax[dh])) / denom[dh];
  const float* hr = h + (long)s * H * C + (long)hh * C + c0;
  float* ar = agg + (long)d * H * C + (long)hh * C + c0;
  float4 v = *(const float4*)hr;
  atomicAdd(ar + 0, v.x * alpha);
  atomicAdd(ar + 1, v.y * alpha);
  atomicAdd(ar + 2, v.z * alpha);
  atomicAdd(ar + 3, v.w * alpha);
}

// ---------- finalize: bias + relu, write next-layer f16 activations ----------
__global__ void k_final12(const float* __restrict__ agg, const float* __restrict__ bias,
                          _Float16* __restrict__ xh, int Nn, int HC) {
  long i = blockIdx.x * (long)blockDim.x + threadIdx.x;
  if (i >= (long)Nn * HC) return;
  int c = (int)(i % HC);
  float v = agg[i] + bias[c];
  xh[i] = (_Float16)(v > 0.f ? v : 0.f);
}

// layer 3: mean over 2 heads + bias + relu -> h3 [N,64]
__global__ void k_final3(const float* __restrict__ agg, const float* __restrict__ bias,
                         float* __restrict__ h3, int Nn) {
  long i = blockIdx.x * (long)blockDim.x + threadIdx.x;
  if (i >= (long)Nn * 64) return;
  long n = i >> 6;
  int c = (int)(i & 63);
  float v = 0.5f * (agg[n * 128 + c] + agg[n * 128 + 64 + c]) + bias[c];
  h3[i] = v > 0.f ? v : 0.f;
}

// ---------- pooling + MLP head ----------
__global__ void k_pool(const float* __restrict__ h3, const int* __restrict__ batch,
                       float* __restrict__ g, int Nn) {
  long i = blockIdx.x * (long)blockDim.x + threadIdx.x;
  if (i >= (long)Nn * 64) return;
  long n = i >> 6;
  int c = (int)(i & 63);
  atomicAdd(&g[(long)batch[n] * 64 + c], h3[i]);
}

__global__ void k_mlp(const float* __restrict__ g, const float* __restrict__ lw1,
                      const float* __restrict__ lb1, const float* __restrict__ lw2,
                      const float* __restrict__ lb2, float* __restrict__ out) {
  __shared__ float sg[64];
  __shared__ float hid[16];
  __shared__ float o[10];
  int b = blockIdx.x, t = threadIdx.x;
  sg[t] = g[(long)b * 64 + t];
  __syncthreads();
  if (t < 16) {
    float s = lb1[t];
    for (int k = 0; k < 64; ++k) s = fmaf(sg[k], lw1[k * 16 + t], s);
    hid[t] = s > 0.f ? s : 0.f;
  }
  __syncthreads();
  if (t < 10) {
    float s = lb2[t];
    for (int k = 0; k < 16; ++k) s = fmaf(hid[k], lw2[k * 10 + t], s);
    o[t] = s;
  }
  __syncthreads();
  if (t == 0) {
    float mx = o[0];
    for (int j = 1; j < 10; ++j) mx = fmaxf(mx, o[j]);
    float sum = 0.f;
    for (int j = 0; j < 10; ++j) sum += __expf(o[j] - mx);
    float lse = mx + __logf(sum);
    for (int j = 0; j < 10; ++j) out[(long)b * 10 + j] = o[j] - lse;
  }
}

extern "C" void kernel_launch(void* const* d_in, const int* in_sizes, int n_in,
                              void* d_out, int out_size, void* d_ws, size_t ws_size,
                              hipStream_t stream) {
  const float* x     = (const float*)d_in[0];
  const int*   ei    = (const int*)d_in[1];
  const int*   batch = (const int*)d_in[2];
  const float* W[3]  = {(const float*)d_in[3],  (const float*)d_in[7],  (const float*)d_in[11]};
  const float* As[3] = {(const float*)d_in[4],  (const float*)d_in[8],  (const float*)d_in[12]};
  const float* Ad[3] = {(const float*)d_in[5],  (const float*)d_in[9],  (const float*)d_in[13]};
  const float* B[3]  = {(const float*)d_in[6],  (const float*)d_in[10], (const float*)d_in[14]};
  const float* lw1 = (const float*)d_in[15];
  const float* lb1 = (const float*)d_in[16];
  const float* lw2 = (const float*)d_in[17];
  const float* lb2 = (const float*)d_in[18];

  const int  Nn = in_sizes[2];      // 100000 nodes
  const int  E  = in_sizes[1] / 2;  // 1.6M directed edges
  const long ET = (long)E + Nn;     // + self loops

  // ---- workspace carve (~167 MB) ----
  char* p = (char*)d_ws;
  auto carve = [&](size_t bytes) -> void* {
    void* r = (void*)p;
    p += (bytes + 255) & ~(size_t)255;
    return r;
  };
  _Float16* xh   = (_Float16*)carve((size_t)Nn * 128 * sizeof(_Float16));  // f16 activations
  _Float16* wt   = (_Float16*)carve((size_t)128 * 128 * sizeof(_Float16)); // f16 W^T
  float*    hw   = (float*)carve((size_t)Nn * 128 * sizeof(float));        // GEMM out
  float*    es   = (float*)carve((size_t)Nn * 8 * sizeof(float));
  float*    ed   = (float*)carve((size_t)Nn * 8 * sizeof(float));
  unsigned* lmax = (unsigned*)carve((size_t)Nn * 8 * sizeof(unsigned));
  float*    den  = (float*)carve((size_t)Nn * 8 * sizeof(float));
  float*    agg  = (float*)carve((size_t)Nn * 128 * sizeof(float));
  float*    h3   = (float*)carve((size_t)Nn * 64 * sizeof(float));
  float*    g    = (float*)carve((size_t)512 * 64 * sizeof(float));

  const int T = 256;  // 8 wave32's per block
  auto blk = [](long n, int t) { return (int)((n + t - 1) / t); };

  const int Kd[3] = {128, 64, 64};
  const int Hh[3] = {8, 8, 2};
  const int Cc[3] = {8, 8, 64};

  k_cvt_f16<<<blk((long)Nn * 128, T), T, 0, stream>>>(x, xh, (long)Nn * 128);

  for (int L = 0; L < 3; ++L) {
    int K = Kd[L], H = Hh[L], C = Cc[L], HC = H * C;

    k_wt<<<blk((long)K * HC, T), T, 0, stream>>>(W[L], wt, K, HC);

    int Mt = Nn / 16;  // 6250 tiles, exact
    int gblocks = blk((long)Mt * 32, T);
    if (HC == 64)
      k_gemm_strip<4><<<gblocks, T, 0, stream>>>(xh, wt, hw, Mt, K);
    else
      k_gemm_strip<8><<<gblocks, T, 0, stream>>>(xh, wt, hw, Mt, K);

    k_attn_coef<<<blk((long)Nn * H, T), T, 0, stream>>>(hw, As[L], Ad[L], es, ed, Nn, H, C);

    k_fill_u32<<<blk((long)Nn * H, T), T, 0, stream>>>(lmax, 0x007FFFFFu, (long)Nn * H); // ord(-inf)
    k_fill_u32<<<blk((long)Nn * H, T), T, 0, stream>>>((unsigned*)den, 0u, (long)Nn * H);
    k_fill_u32<<<blk((long)Nn * HC, T), T, 0, stream>>>((unsigned*)agg, 0u, (long)Nn * HC);

    k_edge_max<<<blk(ET * H, T), T, 0, stream>>>(ei, E, Nn, es, ed, lmax, H);
    k_edge_sum<<<blk(ET * H, T), T, 0, stream>>>(ei, E, Nn, es, ed, lmax, den, H);
    k_edge_agg<<<blk(ET * (HC >> 2), T), T, 0, stream>>>(ei, E, Nn, es, ed, lmax, den, hw, agg, H, C);

    if (L < 2)
      k_final12<<<blk((long)Nn * HC, T), T, 0, stream>>>(agg, B[L], xh, Nn, HC);
    else
      k_final3<<<blk((long)Nn * 64, T), T, 0, stream>>>(agg, B[L], h3, Nn);
  }

  k_fill_u32<<<blk(512 * 64, T), T, 0, stream>>>((unsigned*)g, 0u, 512 * 64);
  k_pool<<<blk((long)Nn * 64, T), T, 0, stream>>>(h3, batch, g, Nn);
  k_mlp<<<512, 64, 0, stream>>>(g, lw1, lb1, lw2, lb2, (float*)d_out);
}
